// RevNN_50354196578780
// MI455X (gfx1250) — matrix operand order repaired and checked by reading, not compiled, not run on hardware
//
#include <hip/hip_runtime.h>

// ---------------------------------------------------------------------------
// RevNN fold on MI455X (gfx1250, wave32).
//  K1 prep:   fp32->bf16 weight conversion + W1x^T packing + barrier reset.
//  K1b xcvt:  fp32->bf16 conversion of both input sequences (packed dwords).
//  K2 gemm:   xproj = x @ W1x^T, v_wmma_f32_16x16x32_bf16, 16x64 strip/wave,
//             fully unrolled K-loop (64 WMMAs per wave).
//  K3 recur:  64 WGs/sequence; bf16 weights TDM-loaded into LDS (40KB/WG) and
//             pinned for all 32767 steps; 3 global barriers per step
//             (+ s_cluster_barrier fast-path hint).
//  K4 logits: 5x1024 matvec + log_softmax.
// ---------------------------------------------------------------------------

#define SEQ      32768
#define WD       512
#define HID      1024
#define NCLS     5
#define NWG      64        // cooperating workgroups per sequence
#define RTHREADS 256

typedef __attribute__((ext_vector_type(16))) __bf16 v16bf;
typedef __attribute__((ext_vector_type(8)))  float  v8f;
typedef unsigned __attribute__((ext_vector_type(4))) v4u;
typedef int      __attribute__((ext_vector_type(4))) v4i;
typedef int      __attribute__((ext_vector_type(8))) v8i;

union Frag16 { v16bf v; unsigned u[8]; };

__device__ __forceinline__ unsigned short f2bf(float f) {
  unsigned u = __float_as_uint(f);
  u += 0x7fffu + ((u >> 16) & 1u);          // round-to-nearest-even
  return (unsigned short)(u >> 16);
}
__device__ __forceinline__ unsigned pk2bf(float lo, float hi) {
  return (unsigned)f2bf(lo) | ((unsigned)f2bf(hi) << 16);
}
__device__ __forceinline__ float bf2f(unsigned h) {
  return __uint_as_float(h << 16);
}

// ---------------------------------------------------------------------------
// TDM: 1-D tile load global->LDS, descriptor per CDNA5 ISA ch.8.
// nelem = element count (2-byte elements), must fit tile_dim0 (16-bit).
// lds_off is the byte offset inside the workgroup's LDS allocation
// (dynamic LDS base == 0 here: kernel has no static __shared__).
// ---------------------------------------------------------------------------
__device__ __forceinline__ void tdm_load_1d(const void* gptr, unsigned lds_off,
                                            unsigned nelem) {
  unsigned long long ga = (unsigned long long)(uintptr_t)gptr;
  v4u g0;
  g0[0] = 1u;                                         // count=1, user descriptor
  g0[1] = lds_off;                                    // lds_addr (bytes)
  g0[2] = (unsigned)(ga & 0xffffffffu);               // global_addr[31:0]
  g0[3] = (unsigned)((ga >> 32) & 0x01ffffffu)        // global_addr[56:32]
        | (2u << 30);                                 // type = 2 ("image")
  v8i g1;
  g1[0] = (int)(1u << 16);                            // data_size=1 (2B), mask=0
  g1[1] = (int)((nelem & 0xffffu) << 16);             // tensor_dim0[15:0]
  g1[2] = (int)((nelem >> 16) & 0xffffu)              // tensor_dim0[31:16]
        | (int)(1u << 16);                            // tensor_dim1 = 1
  g1[3] = (int)((nelem & 0xffffu) << 16);             // tile_dim0 = nelem
  g1[4] = 1;                                          // tile_dim1=1, tile_dim2=0
  g1[5] = (int)nelem;                                 // tensor_dim0_stride lo32
  g1[6] = 0;
  g1[7] = 0;
  v4i z4 = {0, 0, 0, 0};
#if __clang_major__ >= 23
  v8i z8 = {0, 0, 0, 0, 0, 0, 0, 0};
  __builtin_amdgcn_tensor_load_to_lds(g0, g1, z4, z4, z8, 0);
#else
  __builtin_amdgcn_tensor_load_to_lds(g0, g1, z4, z4, 0);
#endif
}

// ---------------------------------------------------------------------------
// K1: weight conversion / packing + barrier reset
// ---------------------------------------------------------------------------
__global__ void __launch_bounds__(256) prep_kernel(
    const float* __restrict__ W1, const float* __restrict__ W2,
    const float* __restrict__ W3,
    unsigned short* __restrict__ w1p, unsigned short* __restrict__ w1xT,
    unsigned short* __restrict__ w2b, unsigned short* __restrict__ w3b,
    int* __restrict__ bar) {
  int i = blockIdx.x * blockDim.x + threadIdx.x;
  if (i < 4) bar[i] = 0;                       // {cnt,phase} x 2 groups
  if (i < WD * WD) {
    int m = i >> 9, k = i & 511;
    w1p[i] = f2bf(W1[m * (2 * WD) + k]);       // W1p row-major (m,k)
    // W1x^T stored [k][n]: K2 packs each WMMA-B dword from pair (k, k+1) at fixed n
    w1xT[(size_t)k * WD + m] = f2bf(W1[m * (2 * WD) + WD + k]);
  }
  if (i < HID * WD) w2b[i] = f2bf(W2[i]);      // (1024,512) row-major
  if (i < WD * HID) w3b[i] = f2bf(W3[i]);      // (512,1024) row-major
}

// ---------------------------------------------------------------------------
// K1b: input fp32 -> packed bf16 dwords (both sequences)
// ---------------------------------------------------------------------------
__global__ void __launch_bounds__(256) xcvt_kernel(
    const float* __restrict__ x0, const float* __restrict__ x1,
    unsigned* __restrict__ xbf) {
  const int i = blockIdx.x * blockDim.x + threadIdx.x;   // dword id, 2^24 total
  const int seq = i >> 23;                               // 2^23 dwords per seq
  const int j = i & 0x7fffff;
  const float* __restrict__ x = seq ? x1 : x0;
  float2 f = *(const float2*)(x + 2 * (size_t)j);
  xbf[i] = pk2bf(f.x, f.y);
}

// ---------------------------------------------------------------------------
// K2: xproj = x @ W1x^T via v_wmma_f32_16x16x32_bf16.
// One wave per 16(s) x 64(n) strip: A fragment reused across 4 B tiles.
// Fully unrolled K loop (16 steps) -> 64 WMMAs per wave.
// Fragment layouts per CDNA5 ISA 7.12.2.
// ---------------------------------------------------------------------------
__global__ void __launch_bounds__(256) xproj_gemm_kernel(
    const unsigned* __restrict__ xbf, const unsigned short* __restrict__ w1xT,
    float* __restrict__ xproj) {
  const int gw   = (blockIdx.x * 256 + threadIdx.x) >> 5;  // global wave id
  const int lane = threadIdx.x & 31;
  const int seq  = gw >> 14;                  // 16384 strips per sequence
  const int rem  = gw & 16383;
  const int s0   = (rem >> 3) << 4;           // 2048 s-tiles
  const int n0   = (rem & 7) << 6;            // 8 strips of 64 columns
  const unsigned* __restrict__ xd = xbf + (size_t)seq * SEQ * (WD / 2);
  float* __restrict__ out = xproj + (size_t)seq * SEQ * WD;
  const unsigned* __restrict__ wq = (const unsigned*)w1xT;

  const int m  = lane & 15;                   // A row / C column
  const int kh = (lane >> 4) * 8;             // A K-half select
  const unsigned* __restrict__ xrow = xd + (size_t)(s0 + m) * (WD / 2);

  const v8f zero = {0.f, 0.f, 0.f, 0.f, 0.f, 0.f, 0.f, 0.f};
  v8f acc[4] = {zero, zero, zero, zero};

#pragma unroll
  for (int kb = 0; kb < WD; kb += 32) {
    Frag16 a;
#pragma unroll
    for (int i = 0; i < 8; ++i) {
      // A(16x32 bf16): VGPR i holds K pair {K0,K0+1}; halves split by lane/16
      const int K0 = (i < 4) ? (kh + 2 * i) : (16 + kh + 2 * (i - 4));
      a.u[i] = xrow[(kb + K0) >> 1];
    }
    // B(32x16 bf16): lane%16 = N col; VGPR i packs K pair (same K order as A); 4 adjacent N tiles
    const unsigned short* wcol = w1xT + (size_t)kb * WD + n0 + m;
#pragma unroll
    for (int t = 0; t < 4; ++t) {
      Frag16 b;
#pragma unroll
      for (int i = 0; i < 8; ++i) {
        const int K0 = (i < 4) ? (kh + 2 * i) : (16 + kh + 2 * (i - 4));
        b.u[i] = (unsigned)wcol[(size_t)K0 * WD + t * 16] |
                 ((unsigned)wcol[(size_t)(K0 + 1) * WD + t * 16] << 16);
      }
      acc[t] = __builtin_amdgcn_wmma_f32_16x16x32_bf16(false, a.v, false, b.v,
                                                       (short)0, acc[t],
                                                       false, false);
    }
  }
#pragma unroll
  for (int t = 0; t < 4; ++t)
#pragma unroll
    for (int r = 0; r < 8; ++r) {
      const int row = ((lane >> 4) << 3) + r; // C: VGPR r -> M = r + 8*(lane/16)
      out[(size_t)(s0 + row) * WD + n0 + t * 16 + m] = acc[t][r];
    }
}

// ---------------------------------------------------------------------------
// Inter-workgroup barrier for one sequence group (NWG blocks).
// Monotonic phase counter; s_cluster_barrier is a NOP unless the dispatch is
// a cluster, in which case waves arrive pre-synced and the spin is ~free.
// ---------------------------------------------------------------------------
__device__ __forceinline__ void group_barrier(int* cnt, int* ph, int& my_phase) {
  __threadfence();                            // release own global stores
  __syncthreads();
  __builtin_amdgcn_s_cluster_barrier();       // s_barrier_signal/-wait -3
  if (threadIdx.x == 0) {
    const int target = ++my_phase;
    if (atomicAdd(cnt, 1) == NWG - 1) {
      atomicExch(cnt, 0);
      __threadfence();
      atomicAdd(ph, 1);
    } else {
      while (__hip_atomic_load(ph, __ATOMIC_ACQUIRE, __HIP_MEMORY_SCOPE_AGENT)
             < target)
        __builtin_amdgcn_s_sleep(2);
    }
  }
  __syncthreads();
  __threadfence();                            // acquire side
}

// ---------------------------------------------------------------------------
// K3: sequential recurrence. blockIdx.x = group*NWG + wg.
// LDS: 8x512 + 16x512 + 8x1024 bf16 weights (40KB, TDM-loaded) + 4KB staging.
// ---------------------------------------------------------------------------
__global__ void __launch_bounds__(RTHREADS) recurrence_kernel(
    const float* __restrict__ x0, const float* __restrict__ x1,
    const float* __restrict__ b1, const float* __restrict__ b2,
    const float* __restrict__ b3,
    const unsigned short* __restrict__ w1p,
    const unsigned short* __restrict__ w2b,
    const unsigned short* __restrict__ w3b,
    const float* __restrict__ xproj, float* __restrict__ act,
    int* __restrict__ bar) {
  const int group = blockIdx.x / NWG;
  const int wg    = blockIdx.x % NWG;
  const int tid   = threadIdx.x;
  const int wave  = tid >> 5, lane = tid & 31;

  extern __shared__ char smem[];
  unsigned short* lw1 = (unsigned short*)smem;     // 8  x 512 bf16  @ off 0
  unsigned short* lw2 = lw1 + 8 * WD;              // 16 x 512 bf16  @ off 8192
  unsigned short* lw3 = lw2 + 16 * WD;             // 8  x 1024 bf16 @ off 24576
  float* vec = (float*)(lw3 + 8 * HID);            // 1024 f32 staging

  float* __restrict__ pbuf = act + group * WD;
  float* __restrict__ abuf = act + 2 * WD + group * WD;
  float* __restrict__ tbuf = act + 4 * WD + group * HID;
  int* cnt = bar + group * 2;
  int* ph  = cnt + 1;
  const float* __restrict__ xin = group ? x1 : x0;
  const float* __restrict__ xp  = xproj + (size_t)group * SEQ * WD;

  // TDM-stage this WG's weight slices into LDS; they stay resident for all
  // 32767 steps. Contiguous row slices -> three 1-D descriptors, wave 0 only.
  if (wave == 0) {
    tdm_load_1d(w1p + (size_t)(wg * 8) * WD,  0u,     8 * WD);
    tdm_load_1d(w2b + (size_t)(wg * 16) * WD, 8192u,  16 * WD);
    tdm_load_1d(w3b + (size_t)(wg * 8) * HID, 24576u, 8 * HID);
    __builtin_amdgcn_s_wait_tensorcnt(0);
  }
  if (wg == 0)
    for (int i = tid; i < WD; i += RTHREADS) pbuf[i] = xin[i];   // p = x[0]
  __syncthreads();

  int my_phase = 0;
  group_barrier(cnt, ph, my_phase);

  for (int s = 1; s < SEQ; ++s) {
    // -- layer 1: a = relu(W1p@p + xproj[s] + b1); rows [wg*8, wg*8+8)
    for (int i = tid; i < WD; i += RTHREADS) vec[i] = pbuf[i];
    __syncthreads();
    {
      const int row = wg * 8 + wave;                 // one row per wave
      const unsigned* w = (const unsigned*)(lw1 + wave * WD) + lane * 8;
      const float* pv = vec + lane * 16;
      float sum = 0.f;
#pragma unroll
      for (int d = 0; d < 8; ++d) {
        unsigned u = w[d];
        sum += bf2f(u & 0xffffu) * pv[2 * d] + bf2f(u >> 16) * pv[2 * d + 1];
      }
#pragma unroll
      for (int off = 16; off; off >>= 1) sum += __shfl_down(sum, off);
      if (lane == 0)
        abuf[row] = fmaxf(sum + xp[(size_t)s * WD + row] + b1[row], 0.f);
    }
    group_barrier(cnt, ph, my_phase);

    // -- layer 2: t = tanh(W2@a + b2); rows [wg*16, wg*16+16)
    for (int i = tid; i < WD; i += RTHREADS) vec[i] = abuf[i];
    __syncthreads();
    {
      const int sub = tid >> 4, l16 = tid & 15;      // 16 threads per row
      const int row = wg * 16 + sub;
      const unsigned* w = (const unsigned*)(lw2 + sub * WD) + l16 * 16;
      const float* av = vec + l16 * 32;
      float sum = 0.f;
#pragma unroll
      for (int d = 0; d < 16; ++d) {
        unsigned u = w[d];
        sum += bf2f(u & 0xffffu) * av[2 * d] + bf2f(u >> 16) * av[2 * d + 1];
      }
#pragma unroll
      for (int off = 8; off; off >>= 1) sum += __shfl_down(sum, off, 16);
      if (l16 == 0) tbuf[row] = tanhf(sum + b2[row]);
    }
    group_barrier(cnt, ph, my_phase);

    // -- layer 3: p = tanh(W3@t + b3); rows [wg*8, wg*8+8)
    for (int i = tid; i < HID; i += RTHREADS) vec[i] = tbuf[i];
    __syncthreads();
    {
      const int row = wg * 8 + wave;
      const unsigned* w = (const unsigned*)(lw3 + wave * HID) + lane * 16;
      const float* tv = vec + lane * 32;
      float sum = 0.f;
#pragma unroll
      for (int d = 0; d < 16; ++d) {
        unsigned u = w[d];
        sum += bf2f(u & 0xffffu) * tv[2 * d] + bf2f(u >> 16) * tv[2 * d + 1];
      }
#pragma unroll
      for (int off = 16; off; off >>= 1) sum += __shfl_down(sum, off);
      if (lane == 0) pbuf[row] = tanhf(sum + b3[row]);
    }
    group_barrier(cnt, ph, my_phase);
  }
}

// ---------------------------------------------------------------------------
// K4: logits = W4 @ [p0;p1] + b4; out = log_softmax(logits). One wave.
// ---------------------------------------------------------------------------
__global__ void logits_kernel(const float* __restrict__ act,
                              const float* __restrict__ W4,
                              const float* __restrict__ b4,
                              float* __restrict__ out) {
  __shared__ float lg[NCLS];
  const int tid = threadIdx.x;                 // 32 threads
  for (int c = 0; c < NCLS; ++c) {
    float s = 0.f;
    for (int j = tid; j < 2 * WD; j += 32) s += W4[c * 2 * WD + j] * act[j];
#pragma unroll
    for (int off = 16; off; off >>= 1) s += __shfl_down(s, off);
    if (tid == 0) lg[c] = s + b4[c];
  }
  __syncthreads();
  if (tid == 0) {
    float m = lg[0];
    for (int c = 1; c < NCLS; ++c) m = fmaxf(m, lg[c]);
    float e = 0.f;
    for (int c = 0; c < NCLS; ++c) e += expf(lg[c] - m);
    float lse = logf(e);
    for (int c = 0; c < NCLS; ++c) out[c] = lg[c] - m - lse;
  }
}

// ---------------------------------------------------------------------------
extern "C" void kernel_launch(void* const* d_in, const int* in_sizes, int n_in,
                              void* d_out, int out_size, void* d_ws,
                              size_t ws_size, hipStream_t stream) {
  (void)in_sizes; (void)n_in; (void)out_size; (void)ws_size;
  const float* x0 = (const float*)d_in[0];
  const float* x1 = (const float*)d_in[1];
  // d_in[2] = word_dim scalar (unused, compile-time constant)
  const float* W1 = (const float*)d_in[3];
  const float* b1 = (const float*)d_in[4];
  const float* W2 = (const float*)d_in[5];
  const float* b2 = (const float*)d_in[6];
  const float* W3 = (const float*)d_in[7];
  const float* b3 = (const float*)d_in[8];
  const float* W4 = (const float*)d_in[9];
  const float* b4 = (const float*)d_in[10];

  // workspace carve-up (~195 MB; xproj fp32 + xbf dominate)
  char* ws = (char*)d_ws;
  unsigned short* w1p  = (unsigned short*)ws; ws += (size_t)WD * WD * 2;
  unsigned short* w1xT = (unsigned short*)ws; ws += (size_t)WD * WD * 2;
  unsigned short* w2b  = (unsigned short*)ws; ws += (size_t)HID * WD * 2;
  unsigned short* w3b  = (unsigned short*)ws; ws += (size_t)WD * HID * 2;
  unsigned* xbf = (unsigned*)ws; ws += (size_t)2 * SEQ * (WD / 2) * 4;
  float* xproj  = (float*)ws;    ws += (size_t)2 * SEQ * WD * 4;
  float* act    = (float*)ws;    ws += (size_t)(2 * WD + 2 * WD + 2 * HID) * 4;
  int*   bar    = (int*)ws;

  prep_kernel<<<(HID * WD + 255) / 256, 256, 0, stream>>>(
      W1, W2, W3, w1p, w1xT, w2b, w3b, bar);

  xcvt_kernel<<<(2 * SEQ * (WD / 2)) / 256, 256, 0, stream>>>(x0, x1, xbf);

  // 2 seqs x 2048 s-tiles x 8 n-strips = 32768 waves; 8 waves/block
  xproj_gemm_kernel<<<32768 / 8, 256, 0, stream>>>(xbf, w1xT, xproj);

  const size_t smem = (size_t)(8 * WD + 16 * WD + 8 * HID) * 2 + HID * 4;
  recurrence_kernel<<<2 * NWG, RTHREADS, smem, stream>>>(
      x0, x1, b1, b2, b3, w1p, w2b, w3b, xproj, act, bar);

  logits_kernel<<<1, 32, 0, stream>>>(act, W4, b4, (float*)d_out);
}
